// ProbSparseAttention_65438121722671
// MI455X (gfx1250) — matrix operand order repaired and checked by reading, not compiled, not run on hardware
//
#include <hip/hip_runtime.h>
#include <math.h>

#define B_   4
#define L_   2048
#define E_   512
#define H_   8
#define D_   64
#define U_   40
#define NTOP 40
#define NEG_INF (-3.402823466e38f)

typedef __attribute__((ext_vector_type(16))) _Float16 v16h;
typedef __attribute__((ext_vector_type(8)))  _Float16 v8h;
typedef __attribute__((ext_vector_type(4)))  _Float16 v4h;
typedef __attribute__((ext_vector_type(8)))  float    v8f;

union V16U { v16h v; v8h h[2]; };

// ---------------------------------------------------------------------------
// K1: f32 -> f16 weight conversion
// ---------------------------------------------------------------------------
__global__ void convert_f16_kernel(const float* __restrict__ src,
                                   _Float16* __restrict__ dst, int n) {
    int i = blockIdx.x * blockDim.x + threadIdx.x;
    if (i < n) dst[i] = (_Float16)src[i];
}

// ---------------------------------------------------------------------------
// K2/K8: WMMA GEMM  C[M=8192 x N=512] = A(f32, MxK) * W(f16, NxK)^T + bias
// MODE 0: out row-major (M x N)   (final projection -> d_out)
// MODE 1: out in (B,H,L,D) layout (q/k/v projections)
//
// Block: 256 threads = 8 waves; block owns a 16-row stripe (m0) and the full
// N=512 (wave w owns n0 = w*64 as four 16x16 f32 accumulators).
// The 16x512 A tile is staged ONCE into LDS as f16 (row pad +8 halfs ->
// conflict-free ds_load_b128 fragment reads); all 8 waves share it.
// B fragments are read directly from the L2-resident f16 weights, with a
// WGP-scope prefetch of the next k-chunk.
// ---------------------------------------------------------------------------
#define APAD   8
#define ASTR   (E_ + APAD)          // 520 halfs per LDS row

template <int MODE>
__global__ __launch_bounds__(256) void wmma_gemm_nt(
    const float* __restrict__ A, const _Float16* __restrict__ W,
    const float* __restrict__ bias, float* __restrict__ out) {
    const int K = E_, N = E_;
    __shared__ _Float16 aT[16 * ASTR];       // 16.6 KB

    const int wave   = threadIdx.x >> 5;
    const int lane   = threadIdx.x & 31;
    const int lanelo = lane & 15;
    const int lanehi = lane >> 4;
    const int m0 = blockIdx.x * 16;
    const int n0 = wave * 64;

    // ---- stage A tile (16 x 512 f32 -> f16) cooperatively, convert once ----
    {
        const float* Ab = A + (size_t)m0 * K;
        for (int i = threadIdx.x; i < 16 * (E_ / 4); i += 256) {
            int row = i >> 7;                 // / 128 float4 per row
            int c4  = (i & 127) << 2;         // column (x4)
            float4 x = *(const float4*)(Ab + (size_t)row * K + c4);
            v4h p = {(_Float16)x.x, (_Float16)x.y, (_Float16)x.z, (_Float16)x.w};
            *(v4h*)(&aT[row * ASTR + c4]) = p;
        }
    }
    __syncthreads();

    v8f acc[4];
#pragma unroll
    for (int s = 0; s < 4; ++s)
#pragma unroll
        for (int r = 0; r < 8; ++r) acc[s][r] = 0.0f;

    const _Float16* wrow = W + (size_t)(n0 + lanelo) * K + 16 * lanehi;

    for (int kc = 0; kc < K; kc += 32) {
        // WGP-scope prefetch of next k-chunk of the weight rows
        if (kc + 32 < K) __builtin_prefetch(wrow + kc + 32, 0, 3);

        // ---- A fragment from LDS: row = lanelo,
        //      halfs[h] -> k = 16*lanehi + h  ----
        V16U af;
        af.h[0] = *(const v8h*)(&aT[lanelo * ASTR + kc + 16 * lanehi]);
        af.h[1] = *(const v8h*)(&aT[lanelo * ASTR + kc + 16 * lanehi + 8]);

        // ---- B fragments from global: col = lanelo, k = 16*lanehi + h ----
#pragma unroll
        for (int s = 0; s < 4; ++s) {
            const v16h* bp = (const v16h*)(wrow + (size_t)s * 16 * K + kc);
            v16h bf = *bp;
            acc[s] = __builtin_amdgcn_wmma_f32_16x16x32_f16(
                false, af.v, false, bf, (short)0, acc[s], false, false);
        }
    }

    // ---- store: C/D layout  m = r + 8*lanehi, n = lanelo ----
#pragma unroll
    for (int s = 0; s < 4; ++s) {
        int n = n0 + s * 16 + lanelo;
        float bv = bias[n];
#pragma unroll
        for (int r = 0; r < 8; ++r) {
            int m = m0 + r + 8 * lanehi;
            float val = acc[s][r] + bv;
            if (MODE == 0) {
                out[(size_t)m * N + n] = val;
            } else {
                int b = m >> 11, l = m & (L_ - 1);
                int h = n >> 6, d = n & 63;
                out[(((size_t)(b * H_ + h)) * L_ + l) * D_ + d] = val;
            }
        }
    }
}

// ---------------------------------------------------------------------------
// K3: sampled QK + sparsity metric M = max_u(q.k_idx) - sum_u(q.k_idx)/L
// one thread per (b,h,l)
// ---------------------------------------------------------------------------
__global__ __launch_bounds__(256) void qk_sample_kernel(
    const float* __restrict__ q, const float* __restrict__ k,
    const int* __restrict__ isamp, float* __restrict__ M) {
    int t = blockIdx.x * blockDim.x + threadIdx.x;
    if (t >= B_ * H_ * L_) return;
    int l  = t & (L_ - 1);
    int bh = t >> 11;
    const float* qr = q + (size_t)t * D_;
    float qreg[D_];
#pragma unroll
    for (int d = 0; d < D_; ++d) qreg[d] = qr[d];
    const float* kb = k + (size_t)bh * L_ * D_;
    const int* ir = isamp + (size_t)l * U_;
    float mx = NEG_INF, sum = 0.0f;
    for (int u = 0; u < U_; ++u) {
        const float* kr = kb + (size_t)ir[u] * D_;
        float dot = 0.0f;
#pragma unroll
        for (int d = 0; d < D_; ++d) dot += qreg[d] * kr[d];
        mx = fmaxf(mx, dot);
        sum += dot;
    }
    M[t] = mx - sum * (1.0f / (float)L_);
}

// ---------------------------------------------------------------------------
// K4: top-40 indices per (b,h) via iterative argmax (32 blocks)
// ---------------------------------------------------------------------------
__global__ __launch_bounds__(256) void topk_kernel(const float* __restrict__ M,
                                                   int* __restrict__ Mtop) {
    __shared__ float sm[L_];
    __shared__ float rv[256];
    __shared__ int   ri[256];
    int bh = blockIdx.x, t = threadIdx.x;
    for (int i = t; i < L_; i += 256) sm[i] = M[(size_t)bh * L_ + i];
    __syncthreads();
    for (int it = 0; it < NTOP; ++it) {
        float best = NEG_INF;
        int bi = 0;
        for (int i = t; i < L_; i += 256) {
            float v = sm[i];
            if (v > best) { best = v; bi = i; }
        }
        rv[t] = best; ri[t] = bi;
        __syncthreads();
        for (int s = 128; s > 0; s >>= 1) {
            if (t < s) {
                float v2 = rv[t + s]; int i2 = ri[t + s];
                if (v2 > rv[t] || (v2 == rv[t] && i2 < ri[t])) {
                    rv[t] = v2; ri[t] = i2;
                }
            }
            __syncthreads();
        }
        if (t == 0) {
            Mtop[bh * NTOP + it] = ri[0];
            sm[ri[0]] = NEG_INF;
        }
        __syncthreads();
    }
}

// ---------------------------------------------------------------------------
// K5: v mean over L per (b,h,d)  (32 blocks)
// ---------------------------------------------------------------------------
__global__ __launch_bounds__(256) void vmean_kernel(const float* __restrict__ v,
                                                    float* __restrict__ vm) {
    __shared__ float part[256];
    int bh = blockIdx.x, t = threadIdx.x;
    int d = t & 63, stripe = t >> 6;
    const float* vb = v + (size_t)bh * L_ * D_;
    float s = 0.0f;
    for (int l = stripe * 512; l < (stripe + 1) * 512; ++l) s += vb[(size_t)l * D_ + d];
    part[t] = s;
    __syncthreads();
    if (t < 64) {
        float tot = part[t] + part[t + 64] + part[t + 128] + part[t + 192];
        vm[bh * D_ + d] = tot * (1.0f / (float)L_);
    }
}

// ---------------------------------------------------------------------------
// K6: context init: ctx[b,l,e] = vmean[b, e]   (vmean flat over (b,h,d) = (b,e))
// ---------------------------------------------------------------------------
__global__ void ctx_init_kernel(const float* __restrict__ vm,
                                float* __restrict__ ctx) {
    int i = blockIdx.x * blockDim.x + threadIdx.x;
    if (i >= B_ * L_ * E_) return;
    int e  = i & (E_ - 1);
    int b  = i >> 20;   // i / (L_*E_)
    ctx[i] = vm[b * E_ + e];
}

// ---------------------------------------------------------------------------
// K7: fused scores -> softmax -> attn@v -> scatter into context
// one block per (b,h,u), u in [0,40)
// ---------------------------------------------------------------------------
__global__ __launch_bounds__(256) void sparse_attn_kernel(
    const float* __restrict__ q, const float* __restrict__ k,
    const float* __restrict__ v, const int* __restrict__ Mtop,
    float* __restrict__ ctx) {
    __shared__ float sc[L_];
    __shared__ float qrow[D_];
    __shared__ float red[256];
    __shared__ float part[256];
    int bh = blockIdx.x / NTOP;
    int u  = blockIdx.x % NTOP;
    int b = bh >> 3, h = bh & 7;
    int t = threadIdx.x;
    int lq = Mtop[bh * NTOP + u];
    if (t < D_) qrow[t] = q[((size_t)bh * L_ + lq) * D_ + t];
    __syncthreads();

    const float* kb = k + (size_t)bh * L_ * D_;
    float lmax = NEG_INF;
    for (int l = t; l < L_; l += 256) {
        const float* kr = kb + (size_t)l * D_;
        float dot = 0.0f;
#pragma unroll
        for (int d2 = 0; d2 < D_; ++d2) dot += qrow[d2] * kr[d2];
        sc[l] = dot;
        lmax = fmaxf(lmax, dot);
    }
    red[t] = lmax;
    __syncthreads();
    for (int s = 128; s > 0; s >>= 1) {
        if (t < s) red[t] = fmaxf(red[t], red[t + s]);
        __syncthreads();
    }
    float gmax = red[0];
    __syncthreads();

    float lsum = 0.0f;
    for (int l = t; l < L_; l += 256) {
        float e = expf(sc[l] - gmax);
        sc[l] = e;
        lsum += e;
    }
    red[t] = lsum;
    __syncthreads();
    for (int s = 128; s > 0; s >>= 1) {
        if (t < s) red[t] += red[t + s];
        __syncthreads();
    }
    float inv = 1.0f / red[0];
    __syncthreads();

    int d = t & 63, stripe = t >> 6;
    const float* vb = v + (size_t)bh * L_ * D_;
    float acc = 0.0f;
    for (int l = stripe * 512; l < (stripe + 1) * 512; ++l)
        acc += sc[l] * vb[(size_t)l * D_ + d];
    part[t] = acc;
    __syncthreads();
    if (t < 64) {
        float tot = (part[t] + part[t + 64] + part[t + 128] + part[t + 192]) * inv;
        ctx[(((size_t)b * L_ + lq) * H_ + h) * D_ + d] = tot;
    }
}

// ---------------------------------------------------------------------------
extern "C" void kernel_launch(void* const* d_in, const int* in_sizes, int n_in,
                              void* d_out, int out_size, void* d_ws, size_t ws_size,
                              hipStream_t stream) {
    const float* query = (const float*)d_in[0];
    const float* key   = (const float*)d_in[1];
    const float* value = (const float*)d_in[2];
    const int*   isamp = (const int*)d_in[3];
    const float* Wq = (const float*)d_in[4];
    const float* bq = (const float*)d_in[5];
    const float* Wk = (const float*)d_in[6];
    const float* bk = (const float*)d_in[7];
    const float* Wv = (const float*)d_in[8];
    const float* bv = (const float*)d_in[9];
    const float* Wo = (const float*)d_in[10];
    const float* bo = (const float*)d_in[11];

    char* ws = (char*)d_ws;
    size_t off = 0;
    auto take = [&](size_t bytes) -> char* {
        char* p = ws + off;
        off = (off + bytes + 255) & ~(size_t)255;
        return p;
    };

    const size_t WELEMS = (size_t)E_ * E_;     // 262144
    _Float16* wq16 = (_Float16*)take(WELEMS * 2);
    _Float16* wk16 = (_Float16*)take(WELEMS * 2);
    _Float16* wv16 = (_Float16*)take(WELEMS * 2);
    _Float16* wo16 = (_Float16*)take(WELEMS * 2);
    float* qp   = (float*)take((size_t)B_ * H_ * L_ * D_ * 4);  // 16 MB
    float* kp   = (float*)take((size_t)B_ * H_ * L_ * D_ * 4);
    float* vp   = (float*)take((size_t)B_ * H_ * L_ * D_ * 4);
    float* Mbuf = (float*)take((size_t)B_ * H_ * L_ * 4);
    int*   Mtop = (int*)take((size_t)B_ * H_ * NTOP * 4);
    float* vmn  = (float*)take((size_t)B_ * H_ * D_ * 4);
    float* ctx  = (float*)take((size_t)B_ * L_ * E_ * 4);       // 16 MB
    float* outp = (float*)d_out;

    // K1: convert weights to f16
    {
        int n = (int)WELEMS;
        int blocks = (n + 255) / 256;
        convert_f16_kernel<<<blocks, 256, 0, stream>>>(Wq, wq16, n);
        convert_f16_kernel<<<blocks, 256, 0, stream>>>(Wk, wk16, n);
        convert_f16_kernel<<<blocks, 256, 0, stream>>>(Wv, wv16, n);
        convert_f16_kernel<<<blocks, 256, 0, stream>>>(Wo, wo16, n);
    }

    // K2: q/k/v projections (WMMA) -> (B,H,L,D)
    const int gemm_blocks = (B_ * L_) / 16;   // 512
    wmma_gemm_nt<1><<<gemm_blocks, 256, 0, stream>>>(query, wq16, bq, qp);
    wmma_gemm_nt<1><<<gemm_blocks, 256, 0, stream>>>(key,   wk16, bk, kp);
    wmma_gemm_nt<1><<<gemm_blocks, 256, 0, stream>>>(value, wv16, bv, vp);

    // K3: sampled QK metric
    qk_sample_kernel<<<(B_ * H_ * L_) / 256, 256, 0, stream>>>(qp, kp, isamp, Mbuf);

    // K4: top-40 per (b,h)
    topk_kernel<<<B_ * H_, 256, 0, stream>>>(Mbuf, Mtop);

    // K5: v mean per (b,h,d)
    vmean_kernel<<<B_ * H_, 256, 0, stream>>>(vp, vmn);

    // K6: context init (broadcast mean)
    ctx_init_kernel<<<(B_ * L_ * E_) / 256, 256, 0, stream>>>(vmn, ctx);

    // K7: sparse attention rows + scatter
    sparse_attn_kernel<<<B_ * H_ * NTOP, 256, 0, stream>>>(qp, kp, vp, Mtop, ctx);

    // K8: output projection (WMMA) -> d_out row-major (B*L, E)
    wmma_gemm_nt<0><<<gemm_blocks, 256, 0, stream>>>(ctx, wo16, bo, outp);
}
